// SerializedAttention_824633721063
// MI455X (gfx1250) — matrix operand order repaired and checked by reading, not compiled, not run on hardware
//
#include <hip/hip_runtime.h>
#include <hip/hip_fp16.h>

typedef _Float16 half_t;
typedef __attribute__((ext_vector_type(8)))  _Float16 v8h;
typedef __attribute__((ext_vector_type(16))) _Float16 v16h;
typedef __attribute__((ext_vector_type(8)))  float    v8f;
typedef __attribute__((ext_vector_type(4)))  int      v4i;

#define POS_BND 16
#define RPE_NUM 33
#define KTS 40   // LDS row stride (halves) for 32-wide K chunks (+8 pad)
#define LQS 72   // 64-wide rows +8 pad
#define LVS 136  // 128-wide rows +8 pad
#define LPS 136

// ---- gfx1250 async global->LDS copy (ASYNCcnt-tracked), with fallback ----
#if defined(__has_builtin)
#if __has_builtin(__builtin_amdgcn_global_load_async_to_lds_b128)
#define ASYNC_LDS 1
#endif
#endif

#ifdef ASYNC_LDS
typedef __attribute__((address_space(1))) v4i* as1_v4i;
typedef __attribute__((address_space(3))) v4i* as3_v4i;
__device__ __forceinline__ void async_ld16(void* lds, const void* g) {
  __builtin_amdgcn_global_load_async_to_lds_b128((as1_v4i)g, (as3_v4i)lds, 0, 0);
}
__device__ __forceinline__ void async_wait0() {
#if __has_builtin(__builtin_amdgcn_s_wait_asynccnt)
  __builtin_amdgcn_s_wait_asynccnt(0);
#else
  asm volatile("s_wait_asynccnt 0x0" ::: "memory");
#endif
}
#endif

__device__ __forceinline__ v8f wmma_f16(v16h a, v16h b, v8f c) {
  return __builtin_amdgcn_wmma_f32_16x16x32_f16(false, a, false, b, (short)0, c,
                                                false, false);
}

__device__ __forceinline__ v8f v8f_zero() {
  v8f z = {0.f, 0.f, 0.f, 0.f, 0.f, 0.f, 0.f, 0.f};
  return z;
}

// A-matrix 16x32 f16 fragment (ISA 7.12.2): lanes<16 hold K {kc..kc+7, kc+16..kc+23},
// lanes>=16 hold K {kc+8..kc+15, kc+24..kc+31}. row = per-lane M row base in LDS.
__device__ __forceinline__ v16h frag_a(const half_t* row, int sel, int kc) {
  v8h lo = *(const v8h*)(row + kc + sel * 8);
  v8h hi = *(const v8h*)(row + kc + 16 + sel * 8);
  return __builtin_shufflevector(lo, hi, 0, 1, 2, 3, 4, 5, 6, 7, 8, 9, 10, 11,
                                 12, 13, 14, 15);
}

// B-matrix 32x16 f16 fragment: lanes<16 hold K {kc..kc+7, kc+16..kc+23}, lanes>=16 hold
// K {kc+8..kc+15, kc+24..kc+31} for column n = lane%16; source stored row-major as N x K.
__device__ __forceinline__ v16h frag_b(const half_t* row, int sel, int kc) {
  v8h lo = *(const v8h*)(row + kc + sel * 8);
  v8h hi = *(const v8h*)(row + kc + 16 + sel * 8);
  return __builtin_shufflevector(lo, hi, 0, 1, 2, 3, 4, 5, 6, 7, 8, 9, 10, 11,
                                 12, 13, 14, 15);
}

__device__ __forceinline__ void cvt16_f32_to_f16(half_t* dst, const float* src) {
  float4 a = ((const float4*)src)[0];
  float4 b = ((const float4*)src)[1];
  float4 c = ((const float4*)src)[2];
  float4 d = ((const float4*)src)[3];
  v8h lo = {(half_t)a.x, (half_t)a.y, (half_t)a.z, (half_t)a.w,
            (half_t)b.x, (half_t)b.y, (half_t)b.z, (half_t)b.w};
  v8h hi = {(half_t)c.x, (half_t)c.y, (half_t)c.z, (half_t)c.w,
            (half_t)d.x, (half_t)d.y, (half_t)d.z, (half_t)d.w};
  *(v8h*)dst = lo;
  *(v8h*)(dst + 8) = hi;
}

__device__ __forceinline__ int clampi(int x) {
  return x < -POS_BND ? -POS_BND : (x > POS_BND ? POS_BND : x);
}

// ---------------------------------------------------------------------------
// Kernel 1: qkv = feat[order] @ w_qkv^T + b_qkv, stored f16 into q/k/v ws.
// q pre-scaled by HEAD^-0.5 = 0.125 (exact in f16). v stored transposed
// (nP, H, HEAD, PATCH) so the PV GEMM B-fragments are contiguous.
// ---------------------------------------------------------------------------
__global__ __launch_bounds__(256) void qkv_gemm_kernel(
    const float* __restrict__ feat, const float* __restrict__ w_qkv,
    const float* __restrict__ b_qkv, const int* __restrict__ order,
    half_t* __restrict__ q_ws, half_t* __restrict__ k_ws,
    half_t* __restrict__ v_ws) {
  __shared__ half_t As[128 * KTS];
  __shared__ half_t Bs[128 * KTS];
  __shared__ int rows[128];

  const int colTile = blockIdx.x;  // 0..11 (tiles 0-3 -> q, 4-7 -> k, 8-11 -> v)
  const int rowBase = blockIdx.y * 128;
  const int colBase = colTile * 128;
  const int tid = threadIdx.x;
  const int lane = tid & 31;
  const int wave = tid >> 5;
  const int sel = lane >> 4;
  const int l16 = lane & 15;

  if (tid < 128) rows[tid] = order[rowBase + tid];

  const int wm = (wave & 3) * 32;   // 4 waves down
  const int wn = (wave >> 2) * 64;  // 2 waves across

  v8f acc[2][4];
#pragma unroll
  for (int i = 0; i < 2; ++i)
#pragma unroll
    for (int j = 0; j < 4; ++j) acc[i][j] = v8f_zero();

  const int fr = tid >> 1;
  const int fc = (tid & 1) * 16;

  for (int kb = 0; kb < 512; kb += 32) {
    __syncthreads();
    cvt16_f32_to_f16(As + fr * KTS + fc,
                     feat + (size_t)rows[fr] * 512 + kb + fc);
    cvt16_f32_to_f16(Bs + fr * KTS + fc,
                     w_qkv + (size_t)(colBase + fr) * 512 + kb + fc);
    __syncthreads();
    v16h af[2], bf[4];
#pragma unroll
    for (int i = 0; i < 2; ++i)
      af[i] = frag_a(As + (wm + i * 16 + l16) * KTS, sel, 0);
#pragma unroll
    for (int j = 0; j < 4; ++j)
      bf[j] = frag_b(Bs + (wn + j * 16 + l16) * KTS, sel, 0);
#pragma unroll
    for (int i = 0; i < 2; ++i)
#pragma unroll
      for (int j = 0; j < 4; ++j) acc[i][j] = wmma_f16(af[i], bf[j], acc[i][j]);
  }

  const int t = colTile >> 2;  // 0=q 1=k 2=v  (uniform: branch hoisted)
  if (t == 0) {
#pragma unroll
    for (int i = 0; i < 2; ++i)
#pragma unroll
      for (int j = 0; j < 4; ++j)
#pragma unroll
        for (int r = 0; r < 8; ++r) {
          const int grow = rowBase + wm + i * 16 + r + 8 * sel;
          const int col = colBase + wn + j * 16 + l16;
          const int hc = col & 511;
          const size_t qidx =
              (((size_t)(grow >> 7) * 8 + (hc >> 6)) * 128 + (grow & 127)) * 64 +
              (hc & 63);
          q_ws[qidx] = (half_t)((acc[i][j][r] + b_qkv[col]) * 0.125f);
        }
  } else if (t == 1) {
#pragma unroll
    for (int i = 0; i < 2; ++i)
#pragma unroll
      for (int j = 0; j < 4; ++j)
#pragma unroll
        for (int r = 0; r < 8; ++r) {
          const int grow = rowBase + wm + i * 16 + r + 8 * sel;
          const int col = colBase + wn + j * 16 + l16;
          const int hc = col & 511;
          const size_t qidx =
              (((size_t)(grow >> 7) * 8 + (hc >> 6)) * 128 + (grow & 127)) * 64 +
              (hc & 63);
          k_ws[qidx] = (half_t)(acc[i][j][r] + b_qkv[col]);
        }
  } else {
#pragma unroll
    for (int i = 0; i < 2; ++i)
#pragma unroll
      for (int j = 0; j < 4; ++j)
#pragma unroll
        for (int r = 0; r < 8; ++r) {
          const int grow = rowBase + wm + i * 16 + r + 8 * sel;
          const int col = colBase + wn + j * 16 + l16;
          const int hc = col & 511;
          v_ws[(((size_t)(grow >> 7) * 8 + (hc >> 6)) * 64 + (hc & 63)) * 128 +
               (grow & 127)] = (half_t)(acc[i][j][r] + b_qkv[col]);
        }
  }
}

// ---------------------------------------------------------------------------
// Kernel 2: per (patch, head) attention. S = QK^T + rpe_bias, softmax, O = PV.
// ---------------------------------------------------------------------------
__global__ __launch_bounds__(256) void attn_kernel(
    const half_t* __restrict__ q_ws, const half_t* __restrict__ k_ws,
    const half_t* __restrict__ v_ws, const int* __restrict__ rel_pos,
    const float* __restrict__ rpe_table, half_t* __restrict__ o_ws) {
  // lvt: [0,17408)   64x136 halves  (V^T, persistent)
  // lq : [17408, +18432) 128x72     (phase 1)
  // lk : [35840, +18432) 128x72     (phase 1)
  // lp : [17408, +34816) 128x136    (phase 2, reuses lq/lk region)
  // lrpe:[54272, +3168)  99x8 f32
  __shared__ __align__(16) unsigned char smem[57440];
  half_t* lvt = (half_t*)smem;
  half_t* lq = (half_t*)(smem + 17408);
  half_t* lk = (half_t*)(smem + 35840);
  half_t* lp = (half_t*)(smem + 17408);
  float* lrpe = (float*)(smem + 54272);

  const int ph = blockIdx.x;  // p*8 + h; same-patch heads adjacent -> rel_pos L2 reuse
  const int p = ph >> 3;
  const int h = ph & 7;
  const int tid = threadIdx.x;
  const int lane = tid & 31;
  const int wave = tid >> 5;
  const int sel = lane >> 4;
  const int l16 = lane & 15;

  const half_t* qb = q_ws + (size_t)ph * (128 * 64);
  const half_t* kb = k_ws + (size_t)ph * (128 * 64);
  const half_t* vb = v_ws + (size_t)ph * (64 * 128);

#ifdef ASYNC_LDS
  // Async DMA into LDS: no VGPR staging, tracked on ASYNCcnt; overlaps with
  // the rpe-table fill below until a single s_wait_asynccnt 0.
  for (int idx = tid; idx < 128 * 8; idx += 256) {
    const int r = idx >> 3, c = (idx & 7) * 8;
    async_ld16(lq + r * LQS + c, qb + r * 64 + c);
    async_ld16(lk + r * LQS + c, kb + r * 64 + c);
  }
  for (int idx = tid; idx < 64 * 16; idx += 256) {
    const int r = idx >> 4, c = (idx & 15) * 8;
    async_ld16(lvt + r * LVS + c, vb + r * 128 + c);
  }
  for (int i = tid; i < 3 * RPE_NUM * 8; i += 256) lrpe[i] = rpe_table[i];
  async_wait0();
#else
  for (int idx = tid; idx < 128 * 8; idx += 256) {
    const int r = idx >> 3, c = (idx & 7) * 8;
    *(v8h*)(lq + r * LQS + c) = *(const v8h*)(qb + r * 64 + c);
    *(v8h*)(lk + r * LQS + c) = *(const v8h*)(kb + r * 64 + c);
  }
  for (int idx = tid; idx < 64 * 16; idx += 256) {
    const int r = idx >> 4, c = (idx & 15) * 8;
    *(v8h*)(lvt + r * LVS + c) = *(const v8h*)(vb + r * 128 + c);
  }
  for (int i = tid; i < 3 * RPE_NUM * 8; i += 256) lrpe[i] = rpe_table[i];
#endif
  __syncthreads();

  const int mBase = wave * 16;  // each wave owns a 16-row query band

  // ---- S = Q K^T (q pre-scaled) ----
  v8f acc[8];
#pragma unroll
  for (int tn = 0; tn < 8; ++tn) acc[tn] = v8f_zero();
  const v16h aq0 = frag_a(lq + (mBase + l16) * LQS, sel, 0);
  const v16h aq1 = frag_a(lq + (mBase + l16) * LQS, sel, 32);
#pragma unroll
  for (int tn = 0; tn < 8; ++tn) {
    v16h b0 = frag_b(lk + (tn * 16 + l16) * LQS, sel, 0);
    v16h b1 = frag_b(lk + (tn * 16 + l16) * LQS, sel, 32);
    acc[tn] = wmma_f16(aq0, b0, acc[tn]);
    acc[tn] = wmma_f16(aq1, b1, acc[tn]);
  }
  __syncthreads();  // done reading lq/lk; lp may overwrite the region

  // ---- RPE bias + row max (row m lives in one 16-lane half: shfl_xor<16) ----
  const int* rp = rel_pos + (size_t)p * (128 * 128 * 3);
  float rowmax[8], rowsum[8], rowinv[8];
#pragma unroll
  for (int r = 0; r < 8; ++r) rowmax[r] = -3.0e38f;
#pragma unroll
  for (int tn = 0; tn < 8; ++tn) {
#pragma unroll
    for (int r = 0; r < 8; ++r) {
      const int m = mBase + r + 8 * sel;
      const int n = tn * 16 + l16;
      const int* e = rp + ((size_t)m * 128 + n) * 3;
      const int i0 = clampi(e[0]) + POS_BND;
      const int i1 = clampi(e[1]) + POS_BND + RPE_NUM;
      const int i2 = clampi(e[2]) + POS_BND + 2 * RPE_NUM;
      const float bias = lrpe[i0 * 8 + h] + lrpe[i1 * 8 + h] + lrpe[i2 * 8 + h];
      const float s = acc[tn][r] + bias;
      acc[tn][r] = s;
      rowmax[r] = fmaxf(rowmax[r], s);
    }
  }
#pragma unroll
  for (int r = 0; r < 8; ++r) {
#pragma unroll
    for (int o = 1; o < 16; o <<= 1)
      rowmax[r] = fmaxf(rowmax[r], __shfl_xor(rowmax[r], o, 32));
    rowsum[r] = 0.0f;
  }
#pragma unroll
  for (int tn = 0; tn < 8; ++tn) {
#pragma unroll
    for (int r = 0; r < 8; ++r) {
      const float pv = __expf(acc[tn][r] - rowmax[r]);
      rowsum[r] += pv;
      lp[(mBase + r + 8 * sel) * LPS + tn * 16 + l16] = (half_t)pv;
    }
  }
#pragma unroll
  for (int r = 0; r < 8; ++r) {
#pragma unroll
    for (int o = 1; o < 16; o <<= 1) rowsum[r] += __shfl_xor(rowsum[r], o, 32);
    rowinv[r] = 1.0f / rowsum[r];
  }
  __syncthreads();

  // ---- O = P V (V stored transposed; 1/rowsum folded into the store) ----
  v8f oacc[4];
#pragma unroll
  for (int tn = 0; tn < 4; ++tn) oacc[tn] = v8f_zero();
#pragma unroll
  for (int kc = 0; kc < 128; kc += 32) {
    v16h ap = frag_a(lp + (mBase + l16) * LPS, sel, kc);
#pragma unroll
    for (int tn = 0; tn < 4; ++tn) {
      v16h bv = frag_b(lvt + (tn * 16 + l16) * LVS, sel, kc);
      oacc[tn] = wmma_f16(ap, bv, oacc[tn]);
    }
  }
  half_t* ob = o_ws + (size_t)p * (128 * 512) + h * 64;
#pragma unroll
  for (int tn = 0; tn < 4; ++tn) {
#pragma unroll
    for (int r = 0; r < 8; ++r) {
      const int m = mBase + r + 8 * sel;
      const int d = tn * 16 + l16;
      ob[(size_t)m * 512 + d] = (half_t)(oacc[tn][r] * rowinv[r]);
    }
  }
}

// ---------------------------------------------------------------------------
// Kernel 3: out = o[inverse] @ w_proj^T + b_proj  (f32 output)
// ---------------------------------------------------------------------------
__global__ __launch_bounds__(256) void proj_gemm_kernel(
    const half_t* __restrict__ o_ws, const float* __restrict__ w_proj,
    const float* __restrict__ b_proj, const int* __restrict__ inverse,
    float* __restrict__ out, int Ntok) {
  __shared__ half_t As[128 * KTS];
  __shared__ half_t Bs[128 * KTS];
  __shared__ int rows[128];

  const int rowBase = blockIdx.y * 128;
  const int colBase = blockIdx.x * 128;
  const int tid = threadIdx.x;
  const int lane = tid & 31;
  const int wave = tid >> 5;
  const int sel = lane >> 4;
  const int l16 = lane & 15;

  if (tid < 128) {
    const int g = rowBase + tid;
    rows[tid] = (g < Ntok) ? inverse[g] : 0;
  }

  const int wm = (wave & 3) * 32;
  const int wn = (wave >> 2) * 64;

  v8f acc[2][4];
#pragma unroll
  for (int i = 0; i < 2; ++i)
#pragma unroll
    for (int j = 0; j < 4; ++j) acc[i][j] = v8f_zero();

  const int fr = tid >> 1;
  const int fc = (tid & 1) * 16;

  for (int kb = 0; kb < 512; kb += 32) {
    __syncthreads();
    {
      const half_t* s = o_ws + (size_t)rows[fr] * 512 + kb + fc;
      half_t* d = As + fr * KTS + fc;
#ifdef ASYNC_LDS
      async_ld16(d, s);       // gathered f16 A-tile via async DMA
      async_ld16(d + 8, s + 8);
#else
      *(v8h*)d = *(const v8h*)s;
      *(v8h*)(d + 8) = *(const v8h*)(s + 8);
#endif
    }
    cvt16_f32_to_f16(Bs + fr * KTS + fc,
                     w_proj + (size_t)(colBase + fr) * 512 + kb + fc);
#ifdef ASYNC_LDS
    async_wait0();
#endif
    __syncthreads();
    v16h af[2], bf[4];
#pragma unroll
    for (int i = 0; i < 2; ++i)
      af[i] = frag_a(As + (wm + i * 16 + l16) * KTS, sel, 0);
#pragma unroll
    for (int j = 0; j < 4; ++j)
      bf[j] = frag_b(Bs + (wn + j * 16 + l16) * KTS, sel, 0);
#pragma unroll
    for (int i = 0; i < 2; ++i)
#pragma unroll
      for (int j = 0; j < 4; ++j) acc[i][j] = wmma_f16(af[i], bf[j], acc[i][j]);
  }

  if (rowBase + 128 <= Ntok) {  // full tile: straight-line stores
#pragma unroll
    for (int i = 0; i < 2; ++i)
#pragma unroll
      for (int j = 0; j < 4; ++j)
#pragma unroll
        for (int r = 0; r < 8; ++r) {
          const int grow = rowBase + wm + i * 16 + r + 8 * sel;
          const int col = colBase + wn + j * 16 + l16;
          out[(size_t)grow * 512 + col] = acc[i][j][r] + b_proj[col];
        }
  } else {
#pragma unroll
    for (int i = 0; i < 2; ++i)
#pragma unroll
      for (int j = 0; j < 4; ++j)
#pragma unroll
        for (int r = 0; r < 8; ++r) {
          const int grow = rowBase + wm + i * 16 + r + 8 * sel;
          const int col = colBase + wn + j * 16 + l16;
          if (grow < Ntok)
            out[(size_t)grow * 512 + col] = acc[i][j][r] + b_proj[col];
        }
  }
}

extern "C" void kernel_launch(void* const* d_in, const int* in_sizes, int n_in,
                              void* d_out, int out_size, void* d_ws,
                              size_t ws_size, hipStream_t stream) {
  const float* feat = (const float*)d_in[0];
  const float* w_qkv = (const float*)d_in[1];
  const float* b_qkv = (const float*)d_in[2];
  const float* w_proj = (const float*)d_in[3];
  const float* b_proj = (const float*)d_in[4];
  const float* rpe = (const float*)d_in[5];
  const int* order = (const int*)d_in[6];
  const int* inverse = (const int*)d_in[7];
  const int* rel_pos = (const int*)d_in[8];
  float* out = (float*)d_out;

  const int Npad = in_sizes[6];  // padded serialized length (multiple of 128)
  const int Ntok = in_sizes[7];  // true token count
  const int nP = Npad / 128;     // number of patches

  // f16 workspace: q, k, v, o  (4 * Npad * 512 halves)
  const size_t perBuf = (size_t)Npad * 512;
  half_t* q_ws = (half_t*)d_ws;
  half_t* k_ws = q_ws + perBuf;
  half_t* v_ws = k_ws + perBuf;
  half_t* o_ws = v_ws + perBuf;

  qkv_gemm_kernel<<<dim3(12, nP), 256, 0, stream>>>(feat, w_qkv, b_qkv, order,
                                                    q_ws, k_ws, v_ws);
  attn_kernel<<<dim3(nP * 8), 256, 0, stream>>>(q_ws, k_ws, v_ws, rel_pos, rpe,
                                                o_ws);
  proj_gemm_kernel<<<dim3(4, (Ntok + 127) / 128), 256, 0, stream>>>(
      o_ws, w_proj, b_proj, inverse, out, Ntok);
}
